// Qwen3NextSparseMoeBlockForEngine_32392643347145
// MI455X (gfx1250) — compile-verified, matrix operand order checked by reading, not run on hardware
//
#include <hip/hip_runtime.h>
#include <hip/hip_bf16.h>

// ---------------------------------------------------------------------------
// Qwen3-Next sparse MoE block for MI455X (gfx1250, wave32, WMMA bf16 path).
// fp32 in HBM (mandatory ~835MB traffic), v_perm_b32-packed bf16 in-register
// for the matrix engine, fp32 WMMA accumulators, fp32 output.
// M-tile = 32 tokens so each fp32 weight fragment feeds two WMMAs.
// ---------------------------------------------------------------------------

typedef __attribute__((ext_vector_type(16))) __bf16   v16bf;
typedef __attribute__((ext_vector_type(8)))  float    v8f;
typedef __attribute__((ext_vector_type(8)))  unsigned v8u;

constexpr int kH    = 2048;   // hidden dim
constexpr int kT    = 2048;   // tokens (B*S)
constexpr int kE    = 64;     // experts
constexpr int kI    = 512;    // expert intermediate
constexpr int kTopK = 8;
constexpr int kM    = 32;     // token tile

// pack two fp32 -> two bf16 (truncate) in ONE v_perm_b32:
// result = { hi[31:24], hi[23:16], lo[31:24], lo[23:16] }
__device__ __forceinline__ unsigned pk2bf(float lo, float hi) {
  return __builtin_amdgcn_perm(__builtin_bit_cast(unsigned, hi),
                               __builtin_bit_cast(unsigned, lo),
                               0x07060302u);
}

__device__ __forceinline__ __bf16 f2bf1(float f) {
  unsigned u = __builtin_bit_cast(unsigned, f);
  return __builtin_bit_cast(__bf16, (unsigned short)(u >> 16));
}

__device__ __forceinline__ float silu(float x) {
  return x / (1.0f + __expf(-x));
}

// 16 consecutive fp32 from global (4x b128) -> bf16 fragment via 8 v_perm.
__device__ __forceinline__ v16bf ld_row16_g(const float* __restrict__ p) {
  const float4* p4 = (const float4*)p;
  float4 a = p4[0], b = p4[1], c = p4[2], d = p4[3];
  v8u r;
  r[0] = pk2bf(a.x, a.y); r[1] = pk2bf(a.z, a.w);
  r[2] = pk2bf(b.x, b.y); r[3] = pk2bf(b.z, b.w);
  r[4] = pk2bf(c.x, c.y); r[5] = pk2bf(c.z, c.w);
  r[6] = pk2bf(d.x, d.y); r[7] = pk2bf(d.z, d.w);
  return __builtin_bit_cast(v16bf, r);
}

__device__ __forceinline__ v8f wmma_bf16(v16bf a, v16bf b, v8f c) {
  // D = A(16x32 bf16) x B(32x16 bf16) + C(16x16 f32)
  return __builtin_amdgcn_wmma_f32_16x16x32_bf16(
      false, a, false, b, (short)0, c, false, false);
}

// ---------------------------------------------------------------------------
// Kernel 0: zero per-expert token counters.
// ---------------------------------------------------------------------------
__global__ void init_counts_kernel(int* __restrict__ counts) {
  if (threadIdx.x < kE) counts[threadIdx.x] = 0;
}

// ---------------------------------------------------------------------------
// Kernel 1: router. 16 tokens per block, 64 threads (thread = expert).
// ---------------------------------------------------------------------------
__global__ __launch_bounds__(64)
void router_kernel(const float* __restrict__ hs,
                   const float* __restrict__ rw,       // [E,H]
                   const float* __restrict__ seg,      // [1,H]
                   int*   __restrict__ counts,         // [E]
                   int*   __restrict__ tok_idx,        // [E,T]
                   float* __restrict__ tok_w,          // [E,T]
                   float* __restrict__ sig) {          // [T]
  __shared__ float sh_chunk[16][128];
  __shared__ float sh_logit[16][kE];

  const int tid = threadIdx.x;     // expert id
  const int t0  = blockIdx.x * 16;

  float acc[16];
#pragma unroll
  for (int m = 0; m < 16; ++m) acc[m] = 0.0f;

  for (int k0 = 0; k0 < kH; k0 += 128) {
    for (int i = tid; i < 16 * 128; i += 64) {
      int m = i >> 7, k = i & 127;
      sh_chunk[m][k] = hs[(size_t)(t0 + m) * kH + k0 + k];
    }
    __syncthreads();
    for (int k = 0; k < 128; ++k) {
      float w = rw[(size_t)tid * kH + k0 + k];
#pragma unroll
      for (int m = 0; m < 16; ++m) acc[m] += sh_chunk[m][k] * w;
    }
    __syncthreads();
  }
#pragma unroll
  for (int m = 0; m < 16; ++m) sh_logit[m][tid] = acc[m];
  __syncthreads();

  if (tid < 16) {
    const int t = t0 + tid;
    // unnormalized softmax (scale cancels after top-k renormalization)
    float mx = -3.0e38f;
    for (int e = 0; e < kE; ++e) mx = fmaxf(mx, sh_logit[tid][e]);
    for (int e = 0; e < kE; ++e)
      sh_logit[tid][e] = __expf(sh_logit[tid][e] - mx);

    int   sel[kTopK];
    float val[kTopK];
    float sum8 = 0.0f;
    for (int j = 0; j < kTopK; ++j) {
      float best = -1.0f; int bi = 0;
      for (int e = 0; e < kE; ++e) {
        float p = sh_logit[tid][e];
        if (p > best) { best = p; bi = e; }
      }
      sh_logit[tid][bi] = -1.0f;
      sel[j] = bi; val[j] = best; sum8 += best;
    }
    const float inv = 1.0f / sum8;
    for (int j = 0; j < kTopK; ++j) {
      int pos = atomicAdd(&counts[sel[j]], 1);
      tok_idx[sel[j] * kT + pos] = t;
      tok_w[sel[j] * kT + pos]   = val[j] * inv;
    }

    // shared-expert sigmoid token gate
    float dot = 0.0f;
    for (int k = 0; k < kH; ++k) dot += hs[(size_t)t * kH + k] * seg[k];
    sig[t] = 1.0f / (1.0f + __expf(-dot));
  }
}

// ---------------------------------------------------------------------------
// Kernel 2: shared expert (SiLU-gated MLP), 32-token tile / block, 8 waves.
// Writes (initializes) d_out with plain stores.
// ---------------------------------------------------------------------------
__global__ __launch_bounds__(256)
void shared_expert_kernel(const float* __restrict__ hs,
                          const float* __restrict__ wg,   // [I,H]
                          const float* __restrict__ wu,   // [I,H]
                          const float* __restrict__ wd,   // [H,I]
                          const float* __restrict__ sig,  // [T]
                          float* __restrict__ out) {
  // one 128KB buffer: phase1 = h tile [32][H] bf16, phase2 = act [32][I] bf16
  __shared__ unsigned sh_u[kM * kH / 2];
  __shared__ float    s_sig[kM];

  __bf16* sh_h = (__bf16*)sh_u;

  const int tid  = threadIdx.x;
  const int lane = tid & 31;
  const int wave = tid >> 5;
  const int t0   = blockIdx.x * kM;

  if (tid < kM) s_sig[tid] = sig[t0 + tid];

  // stage h tile (fp32 -> bf16 via v_perm), b128 loads / b64 LDS stores
  for (int i = tid; i < kM * (kH / 4); i += 256) {
    int m  = i / (kH / 4);
    int c4 = i % (kH / 4);
    float4 v = ((const float4*)(hs + (size_t)(t0 + m) * kH))[c4];
    uint2 u; u.x = pk2bf(v.x, v.y); u.y = pk2bf(v.z, v.w);
    ((uint2*)sh_u)[i] = u;   // i == m*(kH/4) + c4
  }
  __syncthreads();

  const int an_m = lane & 15;           // fragment row
  const int an_k = (lane >> 4) * 16;    // fragment K half

  // ---- gate/up GEMM: [32,H] x [H,I] twice; 32 N-tiles, 4 per wave ----
  v8f accg[2][4], accu[2][4];
  const float* pg[4]; const float* pu[4];
#pragma unroll
  for (int j = 0; j < 4; ++j) {
    accg[0][j] = {}; accg[1][j] = {}; accu[0][j] = {}; accu[1][j] = {};
    int n = (wave + 8 * j) * 16 + (lane & 15);
    pg[j] = wg + (size_t)n * kH + an_k;
    pu[j] = wu + (size_t)n * kH + an_k;
  }
  for (int k0 = 0; k0 < kH; k0 += 32) {
    v16bf a0 = *(const v16bf*)&sh_h[an_m * kH + k0 + an_k];
    v16bf a1 = *(const v16bf*)&sh_h[(an_m + 16) * kH + k0 + an_k];
#pragma unroll
    for (int j = 0; j < 4; ++j) {
      v16bf bg = ld_row16_g(pg[j] + k0);
      v16bf bu = ld_row16_g(pu[j] + k0);
      accg[0][j] = wmma_bf16(a0, bg, accg[0][j]);
      accg[1][j] = wmma_bf16(a1, bg, accg[1][j]);
      accu[0][j] = wmma_bf16(a0, bu, accu[0][j]);
      accu[1][j] = wmma_bf16(a1, bu, accu[1][j]);
    }
  }
  __syncthreads();                       // h tile dead; reuse buffer for act
  __bf16* sh_act = (__bf16*)sh_u;        // [32][I]
#pragma unroll
  for (int ms = 0; ms < 2; ++ms) {
#pragma unroll
    for (int j = 0; j < 4; ++j) {
      int nb = (wave + 8 * j) * 16 + (lane & 15);
#pragma unroll
      for (int r = 0; r < 8; ++r) {
        int m = r + 8 * (lane >> 4) + 16 * ms;
        sh_act[m * kI + nb] = f2bf1(silu(accg[ms][j][r]) * accu[ms][j][r]);
      }
    }
  }
  __syncthreads();

  // ---- down GEMM: [32,I] x [I,H]; 128 N-tiles, 16 per wave ----
  for (int nt = wave; nt < kH / 16; nt += 8) {
    v8f acc0 = {}, acc1 = {};
    const int nn = nt * 16 + (lane & 15);
    const float* pb = wd + (size_t)nn * kI + an_k;
#pragma unroll
    for (int k0 = 0; k0 < kI; k0 += 32) {
      v16bf a0 = *(const v16bf*)&sh_act[an_m * kI + k0 + an_k];
      v16bf a1 = *(const v16bf*)&sh_act[(an_m + 16) * kI + k0 + an_k];
      v16bf b  = ld_row16_g(pb + k0);
      acc0 = wmma_bf16(a0, b, acc0);
      acc1 = wmma_bf16(a1, b, acc1);
    }
#pragma unroll
    for (int r = 0; r < 8; ++r) {
      int m0 = r + 8 * (lane >> 4);
      out[(size_t)(t0 + m0) * kH + nn]      = acc0[r] * s_sig[m0];
      out[(size_t)(t0 + m0 + 16) * kH + nn] = acc1[r] * s_sig[m0 + 16];
    }
  }
}

// ---------------------------------------------------------------------------
// Kernel 3: routed experts. Grid (expert, token-tile); blocks past this
// expert's token count exit immediately. Gathers <=32 routed tokens, runs the
// fused gate_up + down GEMMs in bf16 WMMA, atomically accumulates into d_out.
// ---------------------------------------------------------------------------
__global__ __launch_bounds__(256)
void expert_kernel(const float* __restrict__ hs,
                   const float* __restrict__ gup,   // [E, 2I, H]
                   const float* __restrict__ dwn,   // [E, H, I]
                   const int*   __restrict__ counts,
                   const int*   __restrict__ tok_idx,
                   const float* __restrict__ tok_w,
                   float* __restrict__ out) {
  __shared__ unsigned sh_u[kM * kH / 2];   // 128KB, h tile then act (aliased)
  __shared__ int      s_tok[kM];
  __shared__ float    s_w[kM];

  __bf16* sh_h = (__bf16*)sh_u;

  const int e     = blockIdx.x;
  const int cnt   = counts[e];
  const int start = blockIdx.y * kM;
  if (start >= cnt) return;

  const int tid  = threadIdx.x;
  const int lane = tid & 31;
  const int wave = tid >> 5;

  if (tid < kM) {
    int idx = start + tid;
    int src = idx < cnt ? idx : cnt - 1;
    s_tok[tid] = tok_idx[e * kT + src];
    s_w[tid]   = idx < cnt ? tok_w[e * kT + src] : 0.0f;  // padded rows add 0
  }
  __syncthreads();

  // gather + stage h tile (fp32 -> bf16 via v_perm)
  for (int i = tid; i < kM * (kH / 4); i += 256) {
    int m  = i / (kH / 4);
    int c4 = i % (kH / 4);
    float4 v = ((const float4*)(hs + (size_t)s_tok[m] * kH))[c4];
    uint2 u; u.x = pk2bf(v.x, v.y); u.y = pk2bf(v.z, v.w);
    ((uint2*)sh_u)[i] = u;
  }
  __syncthreads();

  const int an_m = lane & 15;
  const int an_k = (lane >> 4) * 16;
  const float* wbase = gup + (size_t)e * (2 * kI) * kH;

  // ---- fused gate_up GEMM: rows [0,I)=gate, [I,2I)=up ----
  v8f accg[2][4], accu[2][4];
  const float* pg[4]; const float* pu[4];
#pragma unroll
  for (int j = 0; j < 4; ++j) {
    accg[0][j] = {}; accg[1][j] = {}; accu[0][j] = {}; accu[1][j] = {};
    int n = (wave + 8 * j) * 16 + (lane & 15);
    pg[j] = wbase + (size_t)n * kH + an_k;
    pu[j] = wbase + (size_t)(kI + n) * kH + an_k;
  }
  for (int k0 = 0; k0 < kH; k0 += 32) {
    v16bf a0 = *(const v16bf*)&sh_h[an_m * kH + k0 + an_k];
    v16bf a1 = *(const v16bf*)&sh_h[(an_m + 16) * kH + k0 + an_k];
#pragma unroll
    for (int j = 0; j < 4; ++j) {
      v16bf bg = ld_row16_g(pg[j] + k0);
      v16bf bu = ld_row16_g(pu[j] + k0);
      accg[0][j] = wmma_bf16(a0, bg, accg[0][j]);
      accg[1][j] = wmma_bf16(a1, bg, accg[1][j]);
      accu[0][j] = wmma_bf16(a0, bu, accu[0][j]);
      accu[1][j] = wmma_bf16(a1, bu, accu[1][j]);
    }
  }
  __syncthreads();                       // h tile dead; reuse buffer for act
  __bf16* sh_act = (__bf16*)sh_u;        // [32][I]
#pragma unroll
  for (int ms = 0; ms < 2; ++ms) {
#pragma unroll
    for (int j = 0; j < 4; ++j) {
      int nb = (wave + 8 * j) * 16 + (lane & 15);
#pragma unroll
      for (int r = 0; r < 8; ++r) {
        int m = r + 8 * (lane >> 4) + 16 * ms;
        sh_act[m * kI + nb] = f2bf1(silu(accg[ms][j][r]) * accu[ms][j][r]);
      }
    }
  }
  __syncthreads();

  // ---- down GEMM + weighted atomic scatter into d_out ----
  const float* wdbase = dwn + (size_t)e * kH * kI;
  for (int nt = wave; nt < kH / 16; nt += 8) {
    v8f acc0 = {}, acc1 = {};
    const int nn = nt * 16 + (lane & 15);
    const float* pb = wdbase + (size_t)nn * kI + an_k;
#pragma unroll
    for (int k0 = 0; k0 < kI; k0 += 32) {
      v16bf a0 = *(const v16bf*)&sh_act[an_m * kI + k0 + an_k];
      v16bf a1 = *(const v16bf*)&sh_act[(an_m + 16) * kI + k0 + an_k];
      v16bf b  = ld_row16_g(pb + k0);
      acc0 = wmma_bf16(a0, b, acc0);
      acc1 = wmma_bf16(a1, b, acc1);
    }
#pragma unroll
    for (int r = 0; r < 8; ++r) {
      int m0 = r + 8 * (lane >> 4);
      __hip_atomic_fetch_add(&out[(size_t)s_tok[m0] * kH + nn],
                             acc0[r] * s_w[m0],
                             __ATOMIC_RELAXED, __HIP_MEMORY_SCOPE_AGENT);
      __hip_atomic_fetch_add(&out[(size_t)s_tok[m0 + 16] * kH + nn],
                             acc1[r] * s_w[m0 + 16],
                             __ATOMIC_RELAXED, __HIP_MEMORY_SCOPE_AGENT);
    }
  }
}

// ---------------------------------------------------------------------------
// Launch
// ---------------------------------------------------------------------------
extern "C" void kernel_launch(void* const* d_in, const int* in_sizes, int n_in,
                              void* d_out, int out_size, void* d_ws, size_t ws_size,
                              hipStream_t stream) {
  (void)in_sizes; (void)n_in; (void)out_size; (void)ws_size;

  const float* hs  = (const float*)d_in[0];  // [1,T,H]
  const float* rw  = (const float*)d_in[1];  // [E,H]
  const float* gup = (const float*)d_in[2];  // [E,2I,H]
  const float* dwn = (const float*)d_in[3];  // [E,H,I]
  const float* sgw = (const float*)d_in[4];  // [I,H]
  const float* suw = (const float*)d_in[5];  // [I,H]
  const float* sdw = (const float*)d_in[6];  // [H,I]
  const float* seg = (const float*)d_in[7];  // [1,H]
  float* out = (float*)d_out;

  char* ws = (char*)d_ws;
  int*   counts  = (int*)ws;                                     // 64 ints
  int*   tok_idx = (int*)(ws + 256);                             // E*T ints
  float* tok_w   = (float*)(ws + 256 + (size_t)kE * kT * 4);     // E*T floats
  float* sig     = (float*)(ws + 256 + (size_t)2 * kE * kT * 4); // T floats

  init_counts_kernel<<<1, 64, 0, stream>>>(counts);
  router_kernel<<<kT / 16, 64, 0, stream>>>(hs, rw, seg, counts, tok_idx, tok_w, sig);
  shared_expert_kernel<<<kT / kM, 256, 0, stream>>>(hs, sgw, suw, sdw, sig, out);
  expert_kernel<<<dim3(kE, kT / kM), 256, 0, stream>>>(hs, gup, dwn, counts,
                                                       tok_idx, tok_w, out);
}
